// MambaBlock_42296837931101
// MI455X (gfx1250) — compile-verified, matrix operand order checked by reading, not run on hardware
//
#include <hip/hip_runtime.h>
#include <hip/hip_bf16.h>

typedef __attribute__((ext_vector_type(16))) __bf16 v16bf;
typedef __attribute__((ext_vector_type(8)))  __bf16 v8bf;
typedef __attribute__((ext_vector_type(4)))  __bf16 v4bf;
typedef __attribute__((ext_vector_type(8)))  float  v8f;

#define BM 128
#define BN 128
#define BK 32
#define APITCH 40   // bf16 elems per LDS row: 80B rows -> every 8-elem segment 16B aligned

// ---------------------------------------------------------------------------
// fp32-in / fp32-out GEMM, bf16 WMMA compute.
//   C[M,N] = act( betaF*C + A x B + bias[bias sel] )
//   AMODE 0: A contiguous rows (sak==1, 16B-aligned, lda%4==0)  -> float4 loads
//   AMODE 1: A strided (element (m,k) at A[m*lda + k*sak])      -> scalar loads
//   BMODE 0: B[k*ldb+n], aligned     -> float4 loads along n
//   BMODE 1: B[n*ldb+k], aligned     -> float4 loads along k (transposed operand)
//   BMODE 2: B[k*ldb+n], unaligned   -> scalar loads (bounds-guarded if BOUNDS)
//   BOUNDS: guard B column index and store column against N (M,K always full)
//   bias: always a valid pointer (zeros vector for "none");
//   bias_row: 0 -> bias[col], 1 -> bias[row]
//   act: 0 none, 1 silu, 2 softplus  (branch-free select in epilogue)
// ---------------------------------------------------------------------------
template <int AMODE, int BMODE, bool BOUNDS>
__launch_bounds__(256)
__global__ void gemm_bf16_wmma(const float* __restrict__ A, int lda, int sak,
                               const float* __restrict__ B, int ldb,
                               float* __restrict__ C, int ldc,
                               const float* __restrict__ bias, int bias_row,
                               float betaF, int act, int M, int N, int K)
{
    __shared__ __align__(16) __bf16 As[BM * APITCH];
    __shared__ __align__(16) __bf16 Bs[BN * APITCH];   // stored transposed: Bs[n][k]

    const int tid  = threadIdx.x;
    const int lane = tid & 31;
    const int wave = tid >> 5;
    const int wm   = (wave & 3) * 32;   // 4 waves along M
    const int wn   = (wave >> 2) * 64;  // 2 waves along N
    const int half = lane >> 4;
    const int r16  = lane & 15;
    const int klo  = half * 8;          // K-segment select per ISA 16-bit layout

    const int tileM = blockIdx.y * BM;
    const int tileN = blockIdx.x * BN;

    v8f acc[2][4];
#pragma unroll
    for (int i = 0; i < 2; i++)
#pragma unroll
        for (int j = 0; j < 4; j++)
#pragma unroll
            for (int e = 0; e < 8; e++) acc[i][j][e] = 0.f;

    for (int k0 = 0; k0 < K; k0 += BK) {
        // ================= stage A tile [BM x BK] =================
        if (AMODE == 0) {
#pragma unroll
            for (int it = 0; it < 4; it++) {
                int v = tid + it * 256;
                int m = v >> 3, k = (v & 7) * 4;
                const float4 f = *(const float4*)(A + (long)(tileM + m) * lda + (k0 + k));
                v4bf h;
                h[0] = (__bf16)f.x; h[1] = (__bf16)f.y;
                h[2] = (__bf16)f.z; h[3] = (__bf16)f.w;
                *(v4bf*)&As[m * APITCH + k] = h;
            }
        } else {
#pragma unroll
            for (int it = 0; it < 16; it++) {
                int i = tid + it * 256;
                int m = i >> 5, k = i & 31;
                As[m * APITCH + k] =
                    (__bf16)A[(long)(tileM + m) * lda + (long)(k0 + k) * sak];
            }
        }
        // ================= stage B tile transposed into Bs[n][k] =================
        if (BMODE == 0) {
#pragma unroll
            for (int it = 0; it < 4; it++) {
                int v = tid + it * 256;
                int k = v >> 5, n = (v & 31) * 4;
                const float4 f = *(const float4*)(B + (long)(k0 + k) * ldb + (tileN + n));
                Bs[(n + 0) * APITCH + k] = (__bf16)f.x;
                Bs[(n + 1) * APITCH + k] = (__bf16)f.y;
                Bs[(n + 2) * APITCH + k] = (__bf16)f.z;
                Bs[(n + 3) * APITCH + k] = (__bf16)f.w;
            }
        } else if (BMODE == 1) {
#pragma unroll
            for (int it = 0; it < 4; it++) {
                int v = tid + it * 256;
                int n = v >> 3, k = (v & 7) * 4;
                const float4 f = *(const float4*)(B + (long)(tileN + n) * ldb + (k0 + k));
                v4bf h;
                h[0] = (__bf16)f.x; h[1] = (__bf16)f.y;
                h[2] = (__bf16)f.z; h[3] = (__bf16)f.w;
                *(v4bf*)&Bs[n * APITCH + k] = h;
            }
        } else {
#pragma unroll
            for (int it = 0; it < 16; it++) {
                int i = tid + it * 256;
                int k = i >> 7, n = i & 127;
                int gn = tileN + n;
                float v;
                if (BOUNDS)
                    v = (gn < N) ? B[(long)(k0 + k) * ldb + gn] : 0.f;
                else
                    v = B[(long)(k0 + k) * ldb + gn];
                Bs[n * APITCH + k] = (__bf16)v;
            }
        }
        __syncthreads();

        // ---- build fragments: two 16B ds_load_b128 per fragment ----
        v16bf afrag[2], bfrag[4];
#pragma unroll
        for (int i = 0; i < 2; i++) {
            const __bf16* p = &As[(wm + i * 16 + r16) * APITCH];
            v8bf lo = *(const v8bf*)(p + klo);
            v8bf hi = *(const v8bf*)(p + 16 + klo);
            afrag[i] = __builtin_shufflevector(lo, hi, 0,1,2,3,4,5,6,7,8,9,10,11,12,13,14,15);
        }
#pragma unroll
        for (int j = 0; j < 4; j++) {
            const __bf16* p = &Bs[(wn + j * 16 + r16) * APITCH];
            v8bf lo = *(const v8bf*)(p + klo);
            v8bf hi = *(const v8bf*)(p + 16 + klo);
            bfrag[j] = __builtin_shufflevector(lo, hi, 0,1,2,3,4,5,6,7,8,9,10,11,12,13,14,15);
        }

        // ---- 8 WMMAs: 32x64 of output per wave per K-step ----
#pragma unroll
        for (int i = 0; i < 2; i++)
#pragma unroll
            for (int j = 0; j < 4; j++)
                acc[i][j] = __builtin_amdgcn_wmma_f32_16x16x32_bf16(
                    false, afrag[i], false, bfrag[j], (short)0, acc[i][j], false, false);

        __syncthreads();
    }

    // ---- branch-free epilogue: C/D layout => VGPR e: row = e + half*8, col = r16
#pragma unroll
    for (int i = 0; i < 2; i++)
#pragma unroll
        for (int j = 0; j < 4; j++)
#pragma unroll
            for (int e = 0; e < 8; e++) {
                int row = tileM + wm + i * 16 + e + half * 8;
                int col = tileN + wn + j * 16 + r16;
                if (BOUNDS && col >= N) continue;
                long idx = (long)row * ldc + col;
                float v = acc[i][j][e];
                v = fmaf(betaF, C[idx], v);                    // accumulate (betaF 0/1)
                v += bias[bias_row ? row : col];               // zeros vec when unused
                // activation via one exp + selects (no branches)
                float arg   = (act == 1) ? -v : v;
                float ex    = __expf(arg);
                float onep  = 1.f + ex;
                float silu  = v / onep;
                float sp    = __logf(onep);
                float r = v;
                r = (act == 1) ? silu : r;
                r = (act == 2) ? ((v > 20.f) ? v : sp) : r;
                C[idx] = r;
            }
}

// ---------------------------------------------------------------------------
// Zero padding columns of the [1024 x 1026] buffer + the dummy zero-bias vector.
// ---------------------------------------------------------------------------
__global__ void init_pad_and_zeros(float* __restrict__ xm_pad, float* __restrict__ zeros)
{
    int m = blockIdx.x * blockDim.x + threadIdx.x;
    if (m < 1024) {
        xm_pad[(long)m * 1026]        = 0.f;
        xm_pad[(long)m * 1026 + 1025] = 0.f;
        zeros[m]                      = 0.f;
    }
}

// ---------------------------------------------------------------------------
// Selective scan: one thread per channel d, 16 states in registers,
// B/C rows staged through LDS in 32-step chunks. y[l,d] = sum_n h[l,d,n]*C[l,n]
// ---------------------------------------------------------------------------
__launch_bounds__(256)
__global__ void mamba_scan(const float* __restrict__ delta,
                           const float* __restrict__ xc,
                           const float* __restrict__ Bm,
                           const float* __restrict__ Cm,
                           const float* __restrict__ A_log,
                           float* __restrict__ y)
{
    __shared__ float Bs[32 * 16];
    __shared__ float Cs[32 * 16];

    const int tid = threadIdx.x;
    const int d   = blockIdx.x * 256 + tid;

    float Arow[16], h[16];
#pragma unroll
    for (int n = 0; n < 16; n++) {
        Arow[n] = -__expf(A_log[(long)d * 16 + n]);
        h[n] = 0.f;
    }

    for (int c = 0; c < 32; c++) {
        const int l0 = c * 32;
        for (int i = tid; i < 512; i += 256) {
            Bs[i] = Bm[(long)l0 * 16 + i];
            Cs[i] = Cm[(long)l0 * 16 + i];
        }
        __syncthreads();

        for (int li = 0; li < 32; li++) {
            const int l = l0 + li;
            float dl  = delta[(long)l * 1024 + d];
            float xv  = xc[(long)l * 1024 + d];
            float dlx = dl * xv;
            float accy = 0.f;
#pragma unroll
            for (int n = 0; n < 16; n++) {
                float ab = __expf(dl * Arow[n]);
                h[n] = ab * h[n] + dlx * Bs[li * 16 + n];
                accy += h[n] * Cs[li * 16 + n];
            }
            y[(long)l * 1024 + d] = accy;
        }
        __syncthreads();
    }
}

// ---------------------------------------------------------------------------
extern "C" void kernel_launch(void* const* d_in, const int* in_sizes, int n_in,
                              void* d_out, int out_size, void* d_ws, size_t ws_size,
                              hipStream_t stream)
{
    (void)in_sizes; (void)n_in; (void)out_size; (void)ws_size;

    const float* x     = (const float*)d_in[0];   // [2,1024,1024]
    const float* Wp    = (const float*)d_in[1];   // [1024,1024]
    const float* bp    = (const float*)d_in[2];   // [1024]
    const float* convw = (const float*)d_in[3];   // [1024,1024,3]
    const float* convb = (const float*)d_in[4];   // [1024]
    const float* Alog  = (const float*)d_in[5];   // [1024,16]
    const float* Wd    = (const float*)d_in[6];   // [1024,1024]
    const float* Wb    = (const float*)d_in[7];   // [1024,16]
    const float* Wc    = (const float*)d_in[8];   // [1024,16]
    float* out = (float*)d_out;                   // [1024,1024]

    float* ws = (float*)d_ws;
    size_t o = 0;
    float* xm_pad = ws + o; o += 1024 * 1026;     // p[0] padded [1024 x 1026]
    float* xg     = ws + o; o += 1024 * 1024;     // silu(p[1])
    float* xc     = ws + o; o += 1024 * 1024;     // conv output
    float* dl     = ws + o; o += 1024 * 1024;     // delta
    float* Bmat   = ws + o; o += 1024 * 16;
    float* Cmat   = ws + o; o += 1024 * 16;
    float* y      = ws + o; o += 1024 * 1024;     // ssm output
    float* z      = ws + o; o += 1024 * 1024;     // y @ xg^T
    float* zeros  = ws + o; o += 1024;            // dummy zero bias

    dim3 blk(256);
    dim3 gFull(8, 8);   // 1024x1024 output
    dim3 gThin(1, 8);   // 1024x16  output

    init_pad_and_zeros<<<4, 256, 0, stream>>>(xm_pad, zeros);

    // p0 = x[0] @ Wp^T + bp -> xm_pad interior (ldc=1026, col offset +1)
    gemm_bf16_wmma<0, 1, false><<<gFull, blk, 0, stream>>>(
        x, 1024, 1, Wp, 1024, xm_pad + 1, 1026, bp, 0, 0.f, 0, 1024, 1024, 1024);
    // xg = silu(x[1] @ Wp^T + bp)
    gemm_bf16_wmma<0, 1, false><<<gFull, blk, 0, stream>>>(
        x + 1024 * 1024, 1024, 1, Wp, 1024, xg, 1024, bp, 0, 0.f, 1, 1024, 1024, 1024);
    // conv1d = three shifted-tap GEMMs: xc[o,t] = sum_k cw[:,:,k] @ xm_pad[:, t+k]
    gemm_bf16_wmma<1, 2, false><<<gFull, blk, 0, stream>>>(
        convw + 0, 3072, 3, xm_pad + 0, 1026, xc, 1024, zeros, 0, 0.f, 0, 1024, 1024, 1024);
    gemm_bf16_wmma<1, 2, false><<<gFull, blk, 0, stream>>>(
        convw + 1, 3072, 3, xm_pad + 1, 1026, xc, 1024, zeros, 0, 1.f, 0, 1024, 1024, 1024);
    gemm_bf16_wmma<1, 2, false><<<gFull, blk, 0, stream>>>(
        convw + 2, 3072, 3, xm_pad + 2, 1026, xc, 1024, convb, 1, 1.f, 1, 1024, 1024, 1024);
    // delta = softplus(xc @ Wd)
    gemm_bf16_wmma<0, 0, false><<<gFull, blk, 0, stream>>>(
        xc, 1024, 1, Wd, 1024, dl, 1024, zeros, 0, 0.f, 2, 1024, 1024, 1024);
    // B = xc @ W_B ; C = xc @ W_C   (thin, N=16 -> bounds-guarded B/store columns)
    gemm_bf16_wmma<0, 2, true><<<gThin, blk, 0, stream>>>(
        xc, 1024, 1, Wb, 16, Bmat, 16, zeros, 0, 0.f, 0, 1024, 16, 1024);
    gemm_bf16_wmma<0, 2, true><<<gThin, blk, 0, stream>>>(
        xc, 1024, 1, Wc, 16, Cmat, 16, zeros, 0, 0.f, 0, 1024, 16, 1024);
    // selective scan -> y
    mamba_scan<<<4, 256, 0, stream>>>(dl, xc, Bmat, Cmat, Alog, y);
    // z = y @ xg^T
    gemm_bf16_wmma<0, 1, false><<<gFull, blk, 0, stream>>>(
        y, 1024, 1, xg, 1024, z, 1024, zeros, 0, 0.f, 0, 1024, 1024, 1024);
    // out = z @ Wp^T + bp
    gemm_bf16_wmma<0, 1, false><<<gFull, blk, 0, stream>>>(
        z, 1024, 1, Wp, 1024, out, 1024, bp, 0, 0.f, 0, 1024, 1024, 1024);
}
